// MiniAlignDecoder_15436112461893
// MI455X (gfx1250) — matrix operand order repaired and checked by reading, not compiled, not run on hardware
//
#include <hip/hip_runtime.h>
#include <hip/hip_bf16.h>
#include <math.h>

// Problem sizes (fixed by the reference)
#define B_SZ   32
#define S_SZ   1024
#define DM     1024
#define NLAB   8192
#define NH     16
#define NLAYER 4

typedef __attribute__((ext_vector_type(16))) __bf16 v16bf;
typedef __attribute__((ext_vector_type(8)))  float  v8f;
typedef unsigned int u32x4 __attribute__((ext_vector_type(4)));
typedef int          i32x4 __attribute__((ext_vector_type(4)));
typedef int          i32x8 __attribute__((ext_vector_type(8)));

static __device__ inline unsigned short f2bf(float f) {
  unsigned u = __float_as_uint(f);
  unsigned r = u + 0x7FFFu + ((u >> 16) & 1u);   // round-to-nearest-even
  return (unsigned short)(r >> 16);
}
static __device__ inline float bf2f(unsigned short u) {
  return __uint_as_float(((unsigned)u) << 16);
}
// A-matrix frag (16-bit, 16x32): lanes 0-15 hold K 0..7 & 16..23, lanes 16-31 hold K 8..15 & 24..31
static __device__ inline v16bf ldfragA(const unsigned short* p) {
  union { uint4 u[2]; v16bf v; } cv;
  cv.u[0] = *(const uint4*)(p);
  cv.u[1] = *(const uint4*)(p + 16);
  return cv.v;
}
// B-matrix frag (16-bit, 32x16, stored K-major per row): lanes 0-15 hold K 0..7 & 16..23, lanes 16-31 hold K 8..15 & 24..31
static __device__ inline v16bf ldfragB(const unsigned short* p) {
  union { uint4 u[2]; v16bf v; } cv;
  cv.u[0] = *(const uint4*)(p);
  cv.u[1] = *(const uint4*)(p + 16);
  return cv.v;
}
static __device__ inline v8f wmma_bf16(v16bf a, v16bf b, v8f c) {
  return __builtin_amdgcn_wmma_f32_16x16x32_bf16(false, a, false, b, (short)0, c,
                                                 false, false);
}
static __device__ inline float acosh_pos(float x) {   // x >= 1
  return logf(x + sqrtf(x * x - 1.0f));
}

// ---------------------------------------------------------------------------
// TDM: DMA a [rows x 32] bf16 tile (row-major, row stride K elements) from
// global memory into LDS at byte offset lds_off. Issued by the calling wave;
// completion tracked with TENSORcnt. D# layout per CDNA5 ISA 08_async_tensor §8.
// ---------------------------------------------------------------------------
static __device__ inline void tdm_load_tile_bf16(const unsigned short* gptr,
                                                 unsigned lds_off, int M, int K,
                                                 int rows) {
  const unsigned long long ga = (unsigned long long)(const char*)gptr;
  u32x4 g0;
  g0[0] = 1u;                                    // count=1, user-mode descriptor
  g0[1] = lds_off;                               // lds_addr (bytes)
  g0[2] = (unsigned)(ga & 0xFFFFFFFFull);        // global_addr[31:0]
  g0[3] = (unsigned)((ga >> 32) & 0x01FFFFFFull) // global_addr[56:32]
          | 0x80000000u;                         // type=2 ("image")
  i32x8 g1;
  g1[0] = (int)(1u << 16);                       // data_size=1 (2 bytes)
  g1[1] = (int)(((unsigned)K & 0xFFFFu) << 16);  // tensor_dim0[15:0]
  g1[2] = (int)((((unsigned)K >> 16) & 0xFFFFu)  // tensor_dim0[31:16]
          | (((unsigned)M & 0xFFFFu) << 16));    // tensor_dim1[15:0]
  g1[3] = (int)((((unsigned)M >> 16) & 0xFFFFu)  // tensor_dim1[31:16]
          | (32u << 16));                        // tile_dim0 = 32 elements
  g1[4] = (int)(unsigned)rows;                   // tile_dim1 = rows, tile_dim2=0
  g1[5] = K;                                     // tensor_dim0_stride[31:0]
  g1[6] = 0;                                     // stride[47:32] | dim1_stride lo
  g1[7] = 0;
  i32x4 gz = {};                                 // groups 2/3: dims 2..4 unused
#if defined(__clang_major__) && (__clang_major__ >= 23)
  __builtin_amdgcn_tensor_load_to_lds(g0, g1, gz, gz, (i32x8)0, 0);
#else
  __builtin_amdgcn_tensor_load_to_lds(g0, g1, gz, gz, 0);
#endif
}

// ---------------------------------------------------------------------------
// Generic tiled GEMM:  C[M,N](bf16) = act(A[M,K] * B[K,N] + bias)
// A fp32 (converted in LDS staging) or bf16 (staged via the Tensor Data Mover).
// 128x128 block tile, 8 waves of 32x64, K-step 32, v_wmma_f32_16x16x32_bf16.
// ---------------------------------------------------------------------------
template <bool A_BF16, bool RELU>
__global__ __launch_bounds__(256) void gemm_bias_bf16_kernel(
    const void* __restrict__ Aptr, const float* __restrict__ Bm,
    const float* __restrict__ bias, unsigned short* __restrict__ C,
    int M, int N, int K) {
  __shared__ unsigned short As[128 * 32];  // As[m][k]
  __shared__ unsigned short Bt[128 * 32];  // Bt[n][k]
  const int t    = threadIdx.x;
  const int lane = t & 31;
  const int wave = t >> 5;
  const int m0   = blockIdx.y * 128;
  const int n0   = blockIdx.x * 128;
  const int wm   = (wave & 3) * 32;
  const int wn   = (wave >> 2) * 64;
  const int lr   = lane & 15;
  const int half = lane >> 4;
  v8f acc[2][4] = {};
  for (int k0 = 0; k0 < K; k0 += 32) {
    // ---- stage A tile ----
    if (A_BF16) {
      // async DMA of the 128x32 bf16 tile via the Tensor Data Mover (wave 0)
      if (wave == 0) {
        const unsigned short* Ab = (const unsigned short*)Aptr;
        tdm_load_tile_bf16(Ab + (size_t)m0 * K + k0,
                           (unsigned)(unsigned long long)(uintptr_t)As, M, K,
                           128);
      }
    } else {
      const int r  = t >> 3;
      const int kc = (t & 7) * 4;
#pragma unroll
      for (int p = 0; p < 4; ++p) {
        const int row  = r + p * 32;
        const float* Af = (const float*)Aptr;
        const float4 v =
            *(const float4*)(Af + (size_t)(m0 + row) * K + k0 + kc);
        As[row * 32 + kc + 0] = f2bf(v.x);
        As[row * 32 + kc + 1] = f2bf(v.y);
        As[row * 32 + kc + 2] = f2bf(v.z);
        As[row * 32 + kc + 3] = f2bf(v.w);
      }
    }
    // ---- stage B tile transposed: Bt[n][k] (fp32 -> bf16) ----
    {
      const int kr = t >> 5;
      const int nc = (t & 31) * 4;
#pragma unroll
      for (int p = 0; p < 4; ++p) {
        const int k = kr + p * 8;
        const float4 v = *(const float4*)(Bm + (size_t)(k0 + k) * N + n0 + nc);
        Bt[(nc + 0) * 32 + k] = f2bf(v.x);
        Bt[(nc + 1) * 32 + k] = f2bf(v.y);
        Bt[(nc + 2) * 32 + k] = f2bf(v.z);
        Bt[(nc + 3) * 32 + k] = f2bf(v.w);
      }
    }
    if (k0 + 32 < K) {  // hint next B tile toward the caches
      __builtin_prefetch((const char*)(Bm + (size_t)(k0 + 32) * N + n0), 0, 3);
    }
    if (A_BF16) __builtin_amdgcn_s_wait_tensorcnt(0);  // no-op except wave 0
    __syncthreads();
    v16bf afr[2], bfr[4];
#pragma unroll
    for (int i = 0; i < 2; ++i)
      afr[i] = ldfragA(&As[(wm + i * 16 + lr) * 32 + half * 8]);
#pragma unroll
    for (int j = 0; j < 4; ++j)
      bfr[j] = ldfragB(&Bt[(wn + j * 16 + lr) * 32 + half * 8]);
#pragma unroll
    for (int i = 0; i < 2; ++i)
#pragma unroll
      for (int j = 0; j < 4; ++j) acc[i][j] = wmma_bf16(afr[i], bfr[j], acc[i][j]);
    __syncthreads();
  }
  // ---- epilogue: bias (+relu) -> bf16 ----
#pragma unroll
  for (int i = 0; i < 2; ++i) {
#pragma unroll
    for (int j = 0; j < 4; ++j) {
      const int col  = n0 + wn + j * 16 + lr;
      const float bs = bias[col];
#pragma unroll
      for (int r = 0; r < 8; ++r) {
        const int row = m0 + wm + i * 16 + half * 8 + r;
        float v = acc[i][j][r] + bs;
        if (RELU) v = fmaxf(v, 0.0f);
        C[(size_t)row * N + col] = f2bf(v);
      }
    }
  }
}

// ---------------------------------------------------------------------------
// Projection + expmap0:  ek = expmap0(H[524288,64] @ hyp_w[64,64] + hyp_b)
// One wave per 16 rows; A frags straight from global bf16 H; hyp_w in LDS.
// Epilogue: per-row ||v||^2 via shfl_xor over the 16-lane halves; write ek
// (bf16) and |ek|^2 (fp32).
// ---------------------------------------------------------------------------
__global__ __launch_bounds__(256) void proj_expmap_kernel(
    const unsigned short* __restrict__ H, const float* __restrict__ hw,
    const float* __restrict__ hb, unsigned short* __restrict__ ek,
    float* __restrict__ sqe) {
  __shared__ unsigned short Bt[64 * 64];  // Bt[n][k]
  __shared__ float hbs[64];
  const int t    = threadIdx.x;
  const int lane = t & 31;
  const int wave = t >> 5;
#pragma unroll
  for (int p = 0; p < 4; ++p) {
    const int k = p * 16 + (t >> 4);
    const int n = (t & 15) * 4;
    const float4 v = *(const float4*)(hw + k * 64 + n);
    Bt[(n + 0) * 64 + k] = f2bf(v.x);
    Bt[(n + 1) * 64 + k] = f2bf(v.y);
    Bt[(n + 2) * 64 + k] = f2bf(v.z);
    Bt[(n + 3) * 64 + k] = f2bf(v.w);
  }
  if (t < 64) hbs[t] = hb[t];
  __syncthreads();
  const int lr   = lane & 15;
  const int half = lane >> 4;
  const size_t rowbase = (size_t)blockIdx.x * 128 + (size_t)wave * 16;
  const size_t m       = rowbase + lr;
  v8f acc[4] = {};
#pragma unroll
  for (int ks = 0; ks < 64; ks += 32) {
    const v16bf a = ldfragA(H + m * 64 + ks + half * 8);
#pragma unroll
    for (int j = 0; j < 4; ++j) {
      const v16bf b = ldfragB(&Bt[(j * 16 + lr) * 64 + ks + half * 8]);
      acc[j] = wmma_bf16(a, b, acc[j]);
    }
  }
#pragma unroll
  for (int r = 0; r < 8; ++r) {
    float vv[4];
    float sq = 0.0f;
#pragma unroll
    for (int j = 0; j < 4; ++j) {
      vv[j] = acc[j][r] + hbs[j * 16 + lr];
      sq += vv[j] * vv[j];
    }
    sq += __shfl_xor(sq, 1);
    sq += __shfl_xor(sq, 2);
    sq += __shfl_xor(sq, 4);
    sq += __shfl_xor(sq, 8);
    const float nrm = sqrtf(fmaxf(sq, 1e-12f));
    const float sc  = tanhf(nrm) / nrm;
    const size_t row = rowbase + half * 8 + r;  // WMMA C-layout row
#pragma unroll
    for (int j = 0; j < 4; ++j) ek[row * 64 + j * 16 + lr] = f2bf(sc * vv[j]);
    if (lr == 0) sqe[row] = sc * sc * sq;
  }
}

// ---------------------------------------------------------------------------
__global__ __launch_bounds__(256) void rowsq_kernel(const float* __restrict__ x,
                                                    float* __restrict__ out,
                                                    int rows) {
  const int i = blockIdx.x * 256 + threadIdx.x;
  if (i >= rows) return;
  const float4* p = (const float4*)(x + (size_t)i * 64);
  float s = 0.0f;
#pragma unroll
  for (int c = 0; c < 16; ++c) {
    const float4 v = p[c];
    s += v.x * v.x + v.y * v.y + v.z * v.z + v.w * v.w;
  }
  out[i] = s;
}

// ---------------------------------------------------------------------------
static __device__ inline float block_sum(float v, float* part) {
  const int t = threadIdx.x;
  part[t] = v;
  __syncthreads();
  for (int off = 128; off > 0; off >>= 1) {
    if (t < off) part[t] += part[t + off];
    __syncthreads();
  }
  const float r = part[0];
  __syncthreads();
  return r;
}
static __device__ inline float block_max(float v, float* part) {
  const int t = threadIdx.x;
  part[t] = v;
  __syncthreads();
  for (int off = 128; off > 0; off >>= 1) {
    if (t < off) part[t] = fmaxf(part[t], part[t + off]);
    __syncthreads();
  }
  const float r = part[0];
  __syncthreads();
  return r;
}

// ---------------------------------------------------------------------------
// Fused 4-layer HyperbolicPathSearch: one workgroup per (b,h).
// Softmax normalization cancels in the Einstein midpoint, so only
// wg = exp(score-max)*gamma sums are formed. gamma == kh0 == (1+sq)/(1-sq).
// ---------------------------------------------------------------------------
__global__ __launch_bounds__(256) void attn_kernel(
    const unsigned short* __restrict__ ek, const float* __restrict__ sqe,
    const float* __restrict__ label, const int* __restrict__ y,
    const int* __restrict__ mask_text, float* __restrict__ qhv_out,
    float* __restrict__ qh0_out) {
  const int bh = blockIdx.x;
  const int b  = bh >> 4;
  const unsigned short* ekb = ek + (size_t)bh * S_SZ * 64;
  const float* sqb = sqe + (size_t)bh * S_SZ;
  const int* mt    = mask_text + b * S_SZ;
  __shared__ float q[64];
  __shared__ float qv[64];
  __shared__ float wgs[S_SZ];
  __shared__ float part[256];
  __shared__ float midacc[4][64];
  const int t = threadIdx.x;
  if (t < 64) q[t] = label[(size_t)y[b] * 64 + t];
  __syncthreads();
  for (int layer = 0; layer < NLAYER; ++layer) {
    const float sqq = block_sum((t < 64) ? q[t] * q[t] : 0.0f, part);
    const float den = fmaxf(1.0f - sqq, 1e-12f);
    const float qh0 = (1.0f + sqq) / den;
    if (t < 64) qv[t] = 2.0f * q[t] / den;
    __syncthreads();
    // scores for 4 sequence positions per thread
    float sc[4];
    float lmax = -3.0e38f;
#pragma unroll
    for (int i = 0; i < 4; ++i) {
      const int s    = t + i * 256;
      const float se = sqb[s];
      const float de = fmaxf(1.0f - se, 1e-12f);
      const uint4* ep = (const uint4*)(ekb + (size_t)s * 64);
      float dot = 0.0f;
#pragma unroll
      for (int c = 0; c < 8; ++c) {
        const uint4 u   = ep[c];
        const float* qq = &qv[c * 8];
        dot += qq[0] * __uint_as_float(u.x << 16)
             + qq[1] * __uint_as_float(u.x & 0xFFFF0000u)
             + qq[2] * __uint_as_float(u.y << 16)
             + qq[3] * __uint_as_float(u.y & 0xFFFF0000u)
             + qq[4] * __uint_as_float(u.z << 16)
             + qq[5] * __uint_as_float(u.z & 0xFFFF0000u)
             + qq[6] * __uint_as_float(u.w << 16)
             + qq[7] * __uint_as_float(u.w & 0xFFFF0000u);
      }
      const float inner = (2.0f * dot - qh0 * (1.0f + se)) / de;
      const float x     = fmaxf(-inner, 1.0f + 1e-6f);
      sc[i] = (mt[s] > 0) ? -acosh_pos(x) : -1.0e9f;
      lmax  = fmaxf(lmax, sc[i]);
    }
    const float mx = block_max(lmax, part);
    float lwg = 0.0f;
#pragma unroll
    for (int i = 0; i < 4; ++i) {
      const int s     = t + i * 256;
      const float se  = sqb[s];
      const float de  = fmaxf(1.0f - se, 1e-12f);
      const float w   = expf(sc[i] - mx);
      const float wgv = w * (1.0f + se) / de;          // w * gamma
      wgs[s] = wgv * 2.0f / (1.0f + se);               // * Klein scale
      lwg += wgv;
    }
    const float wgsum = block_sum(lwg, part);          // barriers make wgs visible
    // Einstein midpoint numerator: thread t handles dim d over a 256-long s chunk
    const int d = t & 63;
    const int c = t >> 6;
    float acc = 0.0f;
    for (int s = c * 256; s < c * 256 + 256; ++s)
      acc += wgs[s] * bf2f(ekb[(size_t)s * 64 + d]);
    midacc[c][d] = acc;
    __syncthreads();
    if (t < 64)
      q[t] = (midacc[0][t] + midacc[1][t] + midacc[2][t] + midacc[3][t]) / wgsum;
    __syncthreads();
    const float msq = block_sum((t < 64) ? q[t] * q[t] : 0.0f, part);
    const float dd  = 1.0f + sqrtf(fmaxf(1.0f - msq, 1e-12f));
    if (t < 64) q[t] = q[t] / dd;                      // k2p
    __syncthreads();
  }
  const float sqq = block_sum((t < 64) ? q[t] * q[t] : 0.0f, part);
  const float den = fmaxf(1.0f - sqq, 1e-12f);
  if (t < 64) qhv_out[(size_t)bh * 64 + t] = 2.0f * q[t] / den;
  if (t == 0) qh0_out[bh] = (1.0f + sqq) / den;
}

// ---------------------------------------------------------------------------
// Label distance: WMMA GEMM P[512,8192] = Qhv * label^T, fused arccosh
// epilogue writing d in [B, n_label, H] layout.
// ---------------------------------------------------------------------------
__global__ __launch_bounds__(256) void dist_kernel(
    const float* __restrict__ Aq, const float* __restrict__ qh0,
    const float* __restrict__ label, const float* __restrict__ lsq,
    float* __restrict__ dout) {
  __shared__ unsigned short As[128 * 32];
  __shared__ unsigned short Bt[128 * 32];
  const int t    = threadIdx.x;
  const int lane = t & 31;
  const int wave = t >> 5;
  const int m0   = blockIdx.y * 128;
  const int n0   = blockIdx.x * 128;
  const int wm   = (wave & 3) * 32;
  const int wn   = (wave >> 2) * 64;
  const int lr   = lane & 15;
  const int half = lane >> 4;
  v8f acc[2][4] = {};
  for (int k0 = 0; k0 < 64; k0 += 32) {
    const int r  = t >> 3;
    const int kc = (t & 7) * 4;
#pragma unroll
    for (int p = 0; p < 4; ++p) {
      const int row = r + p * 32;
      {
        const float4 v = *(const float4*)(Aq + (size_t)(m0 + row) * 64 + k0 + kc);
        As[row * 32 + kc + 0] = f2bf(v.x);
        As[row * 32 + kc + 1] = f2bf(v.y);
        As[row * 32 + kc + 2] = f2bf(v.z);
        As[row * 32 + kc + 3] = f2bf(v.w);
      }
      {  // label is [N][K] row-major, which is already Bt layout
        const float4 v =
            *(const float4*)(label + (size_t)(n0 + row) * 64 + k0 + kc);
        Bt[row * 32 + kc + 0] = f2bf(v.x);
        Bt[row * 32 + kc + 1] = f2bf(v.y);
        Bt[row * 32 + kc + 2] = f2bf(v.z);
        Bt[row * 32 + kc + 3] = f2bf(v.w);
      }
    }
    __syncthreads();
    v16bf afr[2], bfr[4];
#pragma unroll
    for (int i = 0; i < 2; ++i)
      afr[i] = ldfragA(&As[(wm + i * 16 + lr) * 32 + half * 8]);
#pragma unroll
    for (int j = 0; j < 4; ++j)
      bfr[j] = ldfragB(&Bt[(wn + j * 16 + lr) * 32 + half * 8]);
#pragma unroll
    for (int i = 0; i < 2; ++i)
#pragma unroll
      for (int j = 0; j < 4; ++j) acc[i][j] = wmma_bf16(afr[i], bfr[j], acc[i][j]);
    __syncthreads();
  }
#pragma unroll
  for (int i = 0; i < 2; ++i) {
#pragma unroll
    for (int j = 0; j < 4; ++j) {
      const int col  = n0 + wn + j * 16 + lr;
      const float ls = lsq[col];
      const float dl = fmaxf(1.0f - ls, 1e-12f);
#pragma unroll
      for (int r = 0; r < 8; ++r) {
        const int row     = m0 + wm + i * 16 + half * 8 + r;
        const float inner = (2.0f * acc[i][j][r] - qh0[row] * (1.0f + ls)) / dl;
        const float x     = fmaxf(-inner, 1.0f + 1e-6f);
        const int bb = row >> 4;
        const int hh = row & 15;
        dout[((size_t)bb * NLAB + col) * NH + hh] = acosh_pos(x);
      }
    }
  }
}

// ---------------------------------------------------------------------------
// Aggregator MLP: thread per (b,label); weights cached in LDS.
// ---------------------------------------------------------------------------
__global__ __launch_bounds__(256) void agg_kernel(
    const float* __restrict__ dmat, const float* __restrict__ w1,
    const float* __restrict__ b1, const float* __restrict__ w2,
    const float* __restrict__ b2, const float* __restrict__ w3,
    const float* __restrict__ b3, const float* __restrict__ w4,
    const float* __restrict__ b4, const int* __restrict__ mask_label,
    float* __restrict__ out) {
  __shared__ float W1[16 * 32], B1[32], W2[32 * 32], B2[32], W3[32 * 16], B3[16],
      W4[16], B4v[1];
  const int t = threadIdx.x;
  for (int i = t; i < 512; i += 256) W1[i] = w1[i];
  for (int i = t; i < 1024; i += 256) W2[i] = w2[i];
  for (int i = t; i < 512; i += 256) W3[i] = w3[i];
  if (t < 32) B1[t] = b1[t];
  if (t < 32) B2[t] = b2[t];
  if (t < 16) B3[t] = b3[t];
  if (t < 16) W4[t] = w4[t];
  if (t == 0) B4v[0] = b4[0];
  __syncthreads();
  const size_t idx = (size_t)blockIdx.x * 256 + t;
  float x0[16];
#pragma unroll
  for (int h = 0; h < 16; ++h) x0[h] = dmat[idx * 16 + h];
  float a1[32];
#pragma unroll
  for (int j = 0; j < 32; ++j) a1[j] = B1[j];
#pragma unroll
  for (int h = 0; h < 16; ++h)
#pragma unroll
    for (int j = 0; j < 32; ++j) a1[j] += x0[h] * W1[h * 32 + j];
#pragma unroll
  for (int j = 0; j < 32; ++j) a1[j] = fmaxf(a1[j], 0.0f);
  float a2[32];
#pragma unroll
  for (int j = 0; j < 32; ++j) a2[j] = B2[j];
#pragma unroll
  for (int h = 0; h < 32; ++h)
#pragma unroll
    for (int j = 0; j < 32; ++j) a2[j] += a1[h] * W2[h * 32 + j];
#pragma unroll
  for (int j = 0; j < 32; ++j) a2[j] = fmaxf(a2[j], 0.0f);
  float a3[16];
#pragma unroll
  for (int j = 0; j < 16; ++j) a3[j] = B3[j];
#pragma unroll
  for (int h = 0; h < 32; ++h)
#pragma unroll
    for (int j = 0; j < 16; ++j) a3[j] += a2[h] * W3[h * 16 + j];
  float o = B4v[0];
#pragma unroll
  for (int h = 0; h < 16; ++h) o += fmaxf(a3[h], 0.0f) * W4[h];
  out[idx] = (mask_label[idx] > 0) ? o : -500.0f;
}

// ---------------------------------------------------------------------------
extern "C" void kernel_launch(void* const* d_in, const int* in_sizes, int n_in,
                              void* d_out, int out_size, void* d_ws,
                              size_t ws_size, hipStream_t stream) {
  const float* e        = (const float*)d_in[0];
  const float* label    = (const float*)d_in[1];
  const float* tc_w1    = (const float*)d_in[2];
  const float* tc_b1    = (const float*)d_in[3];
  const float* tc_w2    = (const float*)d_in[4];
  const float* tc_b2    = (const float*)d_in[5];
  const float* hyp_w    = (const float*)d_in[6];
  const float* hyp_b    = (const float*)d_in[7];
  const float* agg_w1   = (const float*)d_in[8];
  const float* agg_b1   = (const float*)d_in[9];
  const float* agg_w2   = (const float*)d_in[10];
  const float* agg_b2   = (const float*)d_in[11];
  const float* agg_w3   = (const float*)d_in[12];
  const float* agg_b3   = (const float*)d_in[13];
  const float* agg_w4   = (const float*)d_in[14];
  const float* agg_b4   = (const float*)d_in[15];
  const int* y          = (const int*)d_in[16];
  const int* mask_text  = (const int*)d_in[17];
  const int* mask_label = (const int*)d_in[18];
  float* out = (float*)d_out;

  // workspace layout (bytes)
  char* w = (char*)d_ws;
  unsigned short* X1 = (unsigned short*)(w);                      // 128 MiB, reused for ek
  unsigned short* Hb = (unsigned short*)(w + (size_t)134217728);  //  64 MiB
  float* sqe  = (float*)(w + (size_t)201326592);                  //   2 MiB
  float* lsq  = (float*)(w + (size_t)203423744);                  //  32 KiB
  float* qhv  = (float*)(w + (size_t)203456512);                  // 128 KiB
  float* qh0  = (float*)(w + (size_t)203587584);                  //   2 KiB
  float* dmat = (float*)(w + (size_t)203589632);                  //  16 MiB
  unsigned short* ekb = X1;  // X1 dead after GEMM2 -> reuse for ek (64 MiB)

  // 1) X1 = relu(e @ tc_w1 + b1)                      [32768,2048] bf16
  gemm_bias_bf16_kernel<false, true>
      <<<dim3(2048 / 128, 32768 / 128), 256, 0, stream>>>(
          (const void*)e, tc_w1, tc_b1, X1, 32768, 2048, 1024);
  // 2) H = X1 @ tc_w2 + b2  (A tile staged by the TDM) [32768,1024] bf16
  gemm_bias_bf16_kernel<true, false>
      <<<dim3(1024 / 128, 32768 / 128), 256, 0, stream>>>(
          (const void*)X1, tc_w2, tc_b2, Hb, 32768, 1024, 2048);
  // 3) ek = expmap0(H @ hyp_w + hyp_b)  (flat reshape => contiguous rows)
  proj_expmap_kernel<<<dim3(524288 / 128), 256, 0, stream>>>(Hb, hyp_w, hyp_b,
                                                             ekb, sqe);
  // 4) label squared norms
  rowsq_kernel<<<dim3(NLAB / 256), 256, 0, stream>>>(label, lsq, NLAB);
  // 5) fused 4-layer hyperbolic attention, one block per (b,h)
  attn_kernel<<<dim3(B_SZ * NH), 256, 0, stream>>>(ekb, sqe, label, y,
                                                   mask_text, qhv, qh0);
  // 6) distances to all labels (WMMA GEMM + arccosh epilogue)
  dist_kernel<<<dim3(NLAB / 128, (B_SZ * NH) / 128), 256, 0, stream>>>(
      qhv, qh0, label, lsq, dmat);
  // 7) aggregator MLP + label mask
  agg_kernel<<<dim3((B_SZ * NLAB) / 256), 256, 0, stream>>>(
      dmat, agg_w1, agg_b1, agg_w2, agg_b2, agg_w3, agg_b3, agg_w4, agg_b4,
      mask_label, out);

  (void)in_sizes; (void)n_in; (void)out_size; (void)ws_size;
}
